// VitAutoencoder_39986145525815
// MI455X (gfx1250) — matrix operand order repaired and checked by reading, not compiled, not run on hardware
//
#include <hip/hip_runtime.h>

#define EMBED   384
#define HEADS   6
#define DH      64
#define NPATCH  196
#define BATCH   32
#define MROWS   (BATCH * NPATCH)   // 6272 rows in the token stream
#define FF      1536
#define QKVW    1152
#define SROWS   208                // 196 padded to 13 * 16
#define SCOLS   224                // 196 padded to 7 * 32 (WMMA K granularity)
#define BH      (BATCH * HEADS)    // 192

typedef __attribute__((ext_vector_type(16))) _Float16 v16h;
typedef __attribute__((ext_vector_type(8)))  _Float16 v8h;
typedef __attribute__((ext_vector_type(8)))  float    v8f;

typedef __attribute__((address_space(3))) char lds_byte;

static __device__ __forceinline__ v8f wmma_f16(v16h a, v16h b, v8f c) {
    // D = A(16x32 f16) * B(32x16 f16) + C(16x16 f32)
    return __builtin_amdgcn_wmma_f32_16x16x32_f16(false, a, false, b, (short)0, c, false, false);
}

static __device__ __forceinline__ float warp_max(float v) {
    #pragma unroll
    for (int o = 16; o > 0; o >>= 1) v = fmaxf(v, __shfl_xor(v, o));
    return v;
}
static __device__ __forceinline__ float warp_sum(float v) {
    #pragma unroll
    for (int o = 16; o > 0; o >>= 1) v += __shfl_xor(v, o);
    return v;
}

// ---------------------------------------------------------------------------
// Weight pack: W[K x N] f32 row-major -> Wt[N x K] f16 (transposed), so B
// fragments are contiguous 32B loads per lane.
// ---------------------------------------------------------------------------
__global__ void pack_w_kernel(const float* __restrict__ W, _Float16* __restrict__ Wt,
                              int K, int N) {
    int idx = blockIdx.x * 256 + threadIdx.x;
    if (idx >= K * N) return;
    int n = idx / K, k = idx % K;
    Wt[idx] = (_Float16)W[(size_t)k * N + n];
}

// ---------------------------------------------------------------------------
// Generic WMMA GEMM with async-to-LDS double-buffered weight staging.
// Block = 4 waves = 4 m-tiles x one 64-col n-block. The 64x64(f16) weight
// chunk shared by all 4 waves is staged into LDS with
// GLOBAL_LOAD_ASYNC_TO_LDS_B128 (ASYNCcnt) while the previous chunk feeds
// v_wmma from ds_load_b128.
// res_mode: 0 none, 1 full residual (res[M x N]), 2 pos-embed (res[(row%196) x N])
// act: 0 none, 1 exact GELU
// ---------------------------------------------------------------------------
__global__ void __launch_bounds__(128)
gemm_wmma_kernel(const float* __restrict__ A, const _Float16* __restrict__ Bt,
                 const float* __restrict__ bias, const float* __restrict__ res,
                 float* __restrict__ out, int M, int N, int K,
                 int res_mode, int act) {
    __shared__ _Float16 Bs[2][64 * 64];      // 2 x 8KB double buffer

    const int tid  = threadIdx.x;
    const int lane = tid & 31;
    const int wave = tid >> 5;
    const int hs   = lane >> 4;    // half-wave select
    const int l15  = lane & 15;
    const int mt   = blockIdx.y * 4 + wave;  // M multiple of 64 => always valid A rows
    const int nb   = blockIdx.x;             // 64-column block
    const int nchunk = K >> 6;               // K is a multiple of 64 for every GEMM here

    // Stage chunk c of the 64-row weight tile into Bs[buf] (8KB = 512 x 16B).
    auto stage = [&](int c, int buf) {
        const _Float16* gbase = Bt + (size_t)nb * 64 * K + (size_t)c * 64;
        #pragma unroll
        for (int it = 0; it < 4; ++it) {
            int idx = it * 128 + tid;            // 0..511
            int r   = idx >> 3;                  // tile row (n within block)
            int seg = idx & 7;                   // 16B segment within 128B row
            const _Float16* g = gbase + (size_t)r * K + seg * 8;
            lds_byte* l = (lds_byte*)&Bs[buf][r * 64 + seg * 8];
            asm volatile("global_load_async_to_lds_b128 %0, %1, off"
                         :: "v"(l), "v"(g) : "memory");
        }
    };

    const float* arow = A + (size_t)(mt * 16 + l15) * K;
    const int o = hs * 16;
    v8f acc[4] = {};

    stage(0, 0);
    for (int c = 0; c < nchunk; ++c) {
        const int buf = c & 1;
        asm volatile("s_wait_asynccnt 0x0" ::: "memory");  // chunk c landed in LDS
        __syncthreads();
        if (c + 1 < nchunk) stage(c + 1, buf ^ 1);

        #pragma unroll
        for (int kk0 = 0; kk0 < 64; kk0 += 32) {
            const int k0 = c * 64 + kk0;
            v16h a;
            #pragma unroll
            for (int i = 0; i < 8; i++) {
                a[i]     = (_Float16)arow[k0 + o + i];
                a[i + 8] = (_Float16)arow[k0 + o + 8 + i];
            }
            #pragma unroll
            for (int j = 0; j < 4; j++) {
                v16h b = *(const v16h*)&Bs[buf][(j * 16 + l15) * 64 + hs * 16 + kk0];
                acc[j] = wmma_f16(a, b, acc[j]);
            }
        }
        __syncthreads();   // all waves done with buf before it is restaged
    }

    #pragma unroll
    for (int j = 0; j < 4; j++) {
        const int col = (nb * 4 + j) * 16 + l15;
        if (col >= N) continue;
        const float bv = bias ? bias[col] : 0.0f;
        #pragma unroll
        for (int r = 0; r < 8; r++) {
            const int row = mt * 16 + r + hs * 8;
            if (row >= M) continue;
            float v = acc[j][r] + bv;
            if (res_mode == 1)      v += res[(size_t)row * N + col];
            else if (res_mode == 2) v += res[(size_t)(row % NPATCH) * N + col];
            if (act == 1) v = 0.5f * v * (1.0f + erff(v * 0.70710678118654752f));
            out[(size_t)row * N + col] = v;
        }
    }
}

// ---------------------------------------------------------------------------
// Attention scores: S[bh, 208, 224] (valid 196x196) = Q * K^T, per (b,h).
// Q rows / K rows gathered from qkv[MROWS x 1152] and converted to f16.
// ---------------------------------------------------------------------------
__global__ void __launch_bounds__(128)
attn_scores_kernel(const float* __restrict__ qkv, float* __restrict__ S) {
    const int lane = threadIdx.x & 31;
    const int wave = threadIdx.x >> 5;
    const int hs   = lane >> 4;
    const int l15  = lane & 15;
    const int mt   = blockIdx.y * 4 + wave;
    if (mt >= 13) return;
    const int nt = blockIdx.x;        // 0..12
    const int bh = blockIdx.z;
    const int b = bh / HEADS, h = bh % HEADS;

    const size_t base = (size_t)b * NPATCH * QKVW;
    int mr = mt * 16 + l15; if (mr > 195) mr = 195;
    int nr = nt * 16 + l15; if (nr > 195) nr = 195;
    const float* qrow = qkv + base + (size_t)mr * QKVW + h * DH;          // Q
    const float* krow = qkv + base + (size_t)nr * QKVW + EMBED + h * DH;  // K

    v8f acc = {};
    const int o = hs * 16;
    #pragma unroll
    for (int k0 = 0; k0 < DH; k0 += 32) {
        v16h a, bm;
        #pragma unroll
        for (int i = 0; i < 8; i++) {
            a[i]     = (_Float16)qrow[k0 + o + i];
            a[i + 8] = (_Float16)qrow[k0 + o + 8 + i];
        }
        const int ko = k0 + hs * 16;
        #pragma unroll
        for (int i = 0; i < 16; i++) bm[i] = (_Float16)krow[ko + i];
        acc = wmma_f16(a, bm, acc);
    }

    float* srow = S + (size_t)bh * SROWS * SCOLS;
    #pragma unroll
    for (int r = 0; r < 8; r++) {
        const int row = mt * 16 + r + hs * 8;       // < 208
        const int col = nt * 16 + l15;              // < 224
        srow[(size_t)row * SCOLS + col] = acc[r];
    }
}

// ---------------------------------------------------------------------------
// Softmax over 196 keys (scale 1/sqrt(64)); writes padded f16 P with zero pad.
// One wave32 per row.
// ---------------------------------------------------------------------------
__global__ void __launch_bounds__(128)
softmax_kernel(const float* __restrict__ S, _Float16* __restrict__ P) {
    const int lane = threadIdx.x & 31;
    const int wave = threadIdx.x >> 5;
    const int m  = blockIdx.x * 4 + wave;  // 0..195
    const int bh = blockIdx.y;
    const float*    srow = S + ((size_t)bh * SROWS + m) * SCOLS;
    _Float16*       prow = P + ((size_t)bh * SROWS + m) * SCOLS;

    const float scale = 0.125f;
    float mx = -3.4e38f;
    for (int j = lane; j < NPATCH; j += 32) mx = fmaxf(mx, srow[j] * scale);
    mx = warp_max(mx);
    float s = 0.0f;
    for (int j = lane; j < NPATCH; j += 32) s += expf(srow[j] * scale - mx);
    s = warp_sum(s);
    const float inv = 1.0f / s;
    for (int j = lane; j < SCOLS; j += 32) {
        float v = (j < NPATCH) ? expf(srow[j] * scale - mx) * inv : 0.0f;
        prow[j] = (_Float16)v;
    }
}

// ---------------------------------------------------------------------------
// V transpose + zero pad: Vt[bh, 64, 224] f16 (Vt[d, k] = V[k, d]).
// ---------------------------------------------------------------------------
__global__ void vtrans_kernel(const float* __restrict__ qkv, _Float16* __restrict__ Vt) {
    int idx = blockIdx.x * 256 + threadIdx.x;
    if (idx >= BH * DH * SCOLS) return;
    int k  = idx % SCOLS;
    int d  = (idx / SCOLS) % DH;
    int bh = idx / (SCOLS * DH);
    int b = bh / HEADS, h = bh % HEADS;
    float v = 0.0f;
    if (k < NPATCH)
        v = qkv[((size_t)(b * NPATCH + k)) * QKVW + 2 * EMBED + h * DH + d];
    Vt[idx] = (_Float16)v;
}

// ---------------------------------------------------------------------------
// O = P (196x224, f16) * V (224x64 via Vt) -> attout[MROWS x 384] per (b,h).
// One wave computes 16 rows x all 64 head dims (4 B fragments).
// ---------------------------------------------------------------------------
__global__ void __launch_bounds__(128)
attn_av_kernel(const _Float16* __restrict__ P, const _Float16* __restrict__ Vt,
               float* __restrict__ out) {
    const int lane = threadIdx.x & 31;
    const int wave = threadIdx.x >> 5;
    const int hs   = lane >> 4;
    const int l15  = lane & 15;
    const int mt   = blockIdx.y * 4 + wave;
    if (mt >= 13) return;
    const int bh = blockIdx.z;
    const int b = bh / HEADS, h = bh % HEADS;

    const _Float16* prow  = P  + ((size_t)bh * SROWS + mt * 16 + l15) * SCOLS;
    const _Float16* vbase = Vt + ((size_t)bh * DH + l15) * SCOLS + hs * 16;

    v8f acc[4] = {};
    const int o = hs * 16;
    #pragma unroll
    for (int k0 = 0; k0 < SCOLS; k0 += 32) {
        v8h lo = *(const v8h*)(prow + k0 + o);
        v8h hi = *(const v8h*)(prow + k0 + o + 8);
        v16h a;
        #pragma unroll
        for (int i = 0; i < 8; i++) { a[i] = lo[i]; a[i + 8] = hi[i]; }
        #pragma unroll
        for (int j = 0; j < 4; j++) {
            v16h bm = *(const v16h*)(vbase + (size_t)(j * 16) * SCOLS + k0);
            acc[j] = wmma_f16(a, bm, acc[j]);
        }
    }

    #pragma unroll
    for (int j = 0; j < 4; j++) {
        #pragma unroll
        for (int r = 0; r < 8; r++) {
            const int row = mt * 16 + r + hs * 8;
            if (row < NPATCH)
                out[((size_t)(b * NPATCH + row)) * EMBED + h * DH + j * 16 + l15] = acc[j][r];
        }
    }
}

// ---------------------------------------------------------------------------
// LayerNorm over D=384: one wave32 per row.
// ---------------------------------------------------------------------------
__global__ void __launch_bounds__(128)
ln_kernel(const float* __restrict__ in, const float* __restrict__ w,
          const float* __restrict__ b, float* __restrict__ out, float eps) {
    const int lane = threadIdx.x & 31;
    const int wave = threadIdx.x >> 5;
    const int row  = blockIdx.x * 4 + wave;
    const float* x = in + (size_t)row * EMBED;
    float s = 0.0f, s2 = 0.0f;
    for (int j = lane; j < EMBED; j += 32) { float v = x[j]; s += v; s2 += v * v; }
    s = warp_sum(s); s2 = warp_sum(s2);
    const float mean = s * (1.0f / EMBED);
    const float var  = s2 * (1.0f / EMBED) - mean * mean;
    const float inv  = rsqrtf(var + eps);
    float* o = out + (size_t)row * EMBED;
    for (int j = lane; j < EMBED; j += 32)
        o[j] = (x[j] - mean) * inv * w[j] + b[j];
}

// ---------------------------------------------------------------------------
// Patchify: x[B,3,224,224] -> xp[B*196, 768], col = c*256 + py*16 + px.
// ---------------------------------------------------------------------------
__global__ void patchify_kernel(const float* __restrict__ x, float* __restrict__ xp) {
    int idx = blockIdx.x * 256 + threadIdx.x;
    if (idx >= MROWS * 768) return;
    int col = idx % 768, row = idx / 768;
    int b = row / NPATCH, pp = row % NPATCH;
    int gy = pp / 14, gx = pp % 14;
    int c = col >> 8, r = col & 255, py = r >> 4, px = r & 15;
    xp[idx] = x[(((size_t)b * 3 + c) * 224 + (gy * 16 + py)) * 224 + gx * 16 + px];
}

// ---------------------------------------------------------------------------
// Fold: rec[B,196,768] -> out[B,3,224,224] with the reference's reshape quirk.
// ---------------------------------------------------------------------------
__global__ void fold_kernel(const float* __restrict__ rec, float* __restrict__ out) {
    int idx = blockIdx.x * 256 + threadIdx.x;
    if (idx >= BATCH * 3 * 224 * 224) return;
    int w  = idx % 224;
    int hh = (idx / 224) % 224;
    int c  = (idx / (224 * 224)) % 3;
    int b  = idx / (3 * 224 * 224);
    int gy = hh >> 4, py = hh & 15;
    int flat = c * 3584 + py * 224 + w;        // within the [14,16,16,3] row block
    int gx = flat / 768;  int rem  = flat % 768;
    int py2 = rem / 48;   int rem2 = rem % 48;
    int px = rem2 / 3;    int c2   = rem2 % 3;
    out[idx] = rec[((size_t)(b * NPATCH) + gy * 14 + gx) * 768 + c2 * 256 + py2 * 16 + px];
}

// ---------------------------------------------------------------------------

extern "C" void kernel_launch(void* const* d_in, const int* in_sizes, int n_in,
                              void* d_out, int out_size, void* d_ws, size_t ws_size,
                              hipStream_t stream) {
    (void)in_sizes; (void)n_in; (void)out_size; (void)ws_size;

    const float* X       = (const float*)d_in[0];
    const float* patch_w = (const float*)d_in[1];
    const float* patch_b = (const float*)d_in[2];
    const float* pos     = (const float*)d_in[3];
    const float* e_ln1w  = (const float*)d_in[4];
    const float* e_ln1b  = (const float*)d_in[5];
    const float* e_qkvw  = (const float*)d_in[6];
    const float* e_qkvb  = (const float*)d_in[7];
    const float* e_projw = (const float*)d_in[8];
    const float* e_projb = (const float*)d_in[9];
    const float* e_ln2w  = (const float*)d_in[10];
    const float* e_ln2b  = (const float*)d_in[11];
    const float* e_fc1w  = (const float*)d_in[12];
    const float* e_fc1b  = (const float*)d_in[13];
    const float* e_fc2w  = (const float*)d_in[14];
    const float* e_fc2b  = (const float*)d_in[15];
    const float* e_nw    = (const float*)d_in[16];
    const float* e_nb    = (const float*)d_in[17];
    const float* demb_w  = (const float*)d_in[18];
    const float* demb_b  = (const float*)d_in[19];
    /* d_in[20] = mask_token, unused (mask all false) */
    const float* d_qkvw  = (const float*)d_in[21];
    const float* d_qkvb  = (const float*)d_in[22];
    const float* d_projw = (const float*)d_in[23];
    const float* d_projb = (const float*)d_in[24];
    const float* d_ln1w  = (const float*)d_in[25];
    const float* d_ln1b  = (const float*)d_in[26];
    const float* d_fc1w  = (const float*)d_in[27];
    const float* d_fc1b  = (const float*)d_in[28];
    const float* d_fc2w  = (const float*)d_in[29];
    const float* d_fc2b  = (const float*)d_in[30];
    const float* d_ln2w  = (const float*)d_in[31];
    const float* d_ln2b  = (const float*)d_in[32];
    const float* d_nw    = (const float*)d_in[33];
    const float* d_nb    = (const float*)d_in[34];
    const float* pred_w  = (const float*)d_in[35];
    const float* pred_b  = (const float*)d_in[36];
    const float* rec_w   = (const float*)d_in[37];
    const float* rec_b   = (const float*)d_in[38];

    // ---- deterministic workspace carving (256B aligned) ----
    char* wsbase = (char*)d_ws;
    size_t off = 0;
    auto alloc = [&](size_t bytes) -> void* {
        void* p = wsbase + off;
        off = (off + bytes + 255) & ~(size_t)255;
        return p;
    };
    auto pack = [&](const float* W, int K, int N) -> const _Float16* {
        _Float16* dst = (_Float16*)alloc((size_t)K * N * sizeof(_Float16));
        int total = K * N;
        pack_w_kernel<<<(total + 255) / 256, 256, 0, stream>>>(W, dst, K, N);
        return dst;
    };

    // pack every weight matrix to transposed fp16 (once per launch)
    const _Float16* pw_t = pack(patch_w, 768, EMBED);
    const _Float16 *eq[12], *ep[12], *e1[12], *e2[12];
    for (int i = 0; i < 12; i++) {
        eq[i] = pack(e_qkvw  + (size_t)i * EMBED * QKVW,  EMBED, QKVW);
        ep[i] = pack(e_projw + (size_t)i * EMBED * EMBED, EMBED, EMBED);
        e1[i] = pack(e_fc1w  + (size_t)i * EMBED * FF,    EMBED, FF);
        e2[i] = pack(e_fc2w  + (size_t)i * FF * EMBED,    FF,    EMBED);
    }
    const _Float16* demb_t = pack(demb_w, EMBED, EMBED);
    const _Float16 *dq[8], *dp[8], *dm1[8], *dm2[8];
    for (int i = 0; i < 8; i++) {
        dq[i]  = pack(d_qkvw  + (size_t)i * EMBED * QKVW,  EMBED, QKVW);
        dp[i]  = pack(d_projw + (size_t)i * EMBED * EMBED, EMBED, EMBED);
        dm1[i] = pack(d_fc1w  + (size_t)i * EMBED * FF,    EMBED, FF);
        dm2[i] = pack(d_fc2w  + (size_t)i * FF * EMBED,    FF,    EMBED);
    }
    const _Float16* pred_t = pack(pred_w, EMBED, EMBED);
    const _Float16* rec_t  = pack(rec_w,  EMBED, 768);

    // activation buffers ("big" is shared by patchified input / MLP hidden / rec)
    float*     big = (float*)alloc((size_t)MROWS * FF * 4);
    float*     t   = (float*)alloc((size_t)MROWS * EMBED * 4);
    float*     tln = (float*)alloc((size_t)MROWS * EMBED * 4);
    float*     u   = (float*)alloc((size_t)MROWS * EMBED * 4);
    float*     qkv = (float*)alloc((size_t)MROWS * QKVW * 4);
    float*     att = (float*)alloc((size_t)MROWS * EMBED * 4);
    float*     S   = (float*)alloc((size_t)BH * SROWS * SCOLS * 4);
    _Float16*  P   = (_Float16*)alloc((size_t)BH * SROWS * SCOLS * 2);
    _Float16*  Vt  = (_Float16*)alloc((size_t)BH * DH * SCOLS * 2);

    auto gemm = [&](const float* A, const _Float16* Bt, const float* bias,
                    const float* res, int res_mode, int act, float* out, int N, int K) {
        dim3 grid(N / 64, (MROWS + 63) / 64);
        gemm_wmma_kernel<<<grid, 128, 0, stream>>>(A, Bt, bias, res, out,
                                                   MROWS, N, K, res_mode, act);
    };
    auto attention = [&](const float* qkvbuf, float* attout) {
        attn_scores_kernel<<<dim3(13, 4, BH), 128, 0, stream>>>(qkvbuf, S);
        softmax_kernel<<<dim3(49, BH), 128, 0, stream>>>(S, P);
        vtrans_kernel<<<(BH * DH * SCOLS + 255) / 256, 256, 0, stream>>>(qkvbuf, Vt);
        attn_av_kernel<<<dim3(1, 4, BH), 128, 0, stream>>>(P, Vt, attout);
    };
    auto ln = [&](const float* in, const float* w, const float* b, float* out, float eps) {
        ln_kernel<<<MROWS / 4, 128, 0, stream>>>(in, w, b, out, eps);
    };

    // ---- patch embed + pos ----
    patchify_kernel<<<(MROWS * 768 + 255) / 256, 256, 0, stream>>>(X, big);
    gemm(big, pw_t, patch_b, pos, /*res_mode=*/2, 0, t, EMBED, 768);

    // ---- encoder: pre-norm blocks, eps=1e-6 ----
    for (int i = 0; i < 12; i++) {
        ln(t, e_ln1w + i * EMBED, e_ln1b + i * EMBED, tln, 1e-6f);
        gemm(tln, eq[i], e_qkvb + i * QKVW, nullptr, 0, 0, qkv, QKVW, EMBED);
        attention(qkv, att);
        gemm(att, ep[i], e_projb + i * EMBED, t, 1, 0, t, EMBED, EMBED);
        ln(t, e_ln2w + i * EMBED, e_ln2b + i * EMBED, tln, 1e-6f);
        gemm(tln, e1[i], e_fc1b + i * FF, nullptr, 0, /*gelu=*/1, big, FF, EMBED);
        gemm(big, e2[i], e_fc2b + i * EMBED, t, 1, 0, t, EMBED, FF);
    }
    ln(t, e_nw, e_nb, tln, 1e-6f);

    // ---- decoder embed (mask all-false => passthrough) ----
    gemm(tln, demb_t, demb_b, nullptr, 0, 0, t, EMBED, EMBED);

    // ---- decoder: post-norm blocks, eps=1e-5 ----
    for (int i = 0; i < 8; i++) {
        gemm(t, dq[i], d_qkvb + i * QKVW, nullptr, 0, 0, qkv, QKVW, EMBED);
        attention(qkv, att);
        gemm(att, dp[i], d_projb + i * EMBED, t, 1, 0, u, EMBED, EMBED);
        ln(u, d_ln1w + i * EMBED, d_ln1b + i * EMBED, t, 1e-5f);
        gemm(t, dm1[i], d_fc1b + i * FF, nullptr, 0, /*gelu=*/1, big, FF, EMBED);
        gemm(big, dm2[i], d_fc2b + i * EMBED, t, 1, 0, u, EMBED, FF);
        ln(u, d_ln2w + i * EMBED, d_ln2b + i * EMBED, t, 1e-5f);
    }
    ln(t, d_nw, d_nb, tln, 1e-5f);

    // ---- pred + reconstruction + fold ----
    gemm(tln, pred_t, pred_b, nullptr, 0, 0, u, EMBED, EMBED);
    gemm(u, rec_t, rec_b, nullptr, 0, 0, big, 768, EMBED);
    fold_kernel<<<(BATCH * 3 * 224 * 224 + 255) / 256, 256, 0, stream>>>(big, (float*)d_out);
}
